// MambaBlock_15607911153877
// MI455X (gfx1250) — matrix operand order repaired and checked by reading, not compiled, not run on hardware
//
#include <hip/hip_runtime.h>
#include <hip/hip_bf16.h>
#include <math.h>

// ---------------------------------------------------------------------------
// Types for CDNA5 WMMA (wave32, 16x16x32 bf16 -> f32)
// ---------------------------------------------------------------------------
typedef __attribute__((ext_vector_type(16))) __bf16          v16bf;
typedef __attribute__((ext_vector_type(16))) unsigned short  v16us;
typedef __attribute__((ext_vector_type(8)))  float           v8f;

#define BATCH 32
#define SEQ   2048
#define HID   512
#define EXP   1024
#define STATE 16
#define NTOK  (BATCH * SEQ)   // 65536

// float -> bf16 bits, round-to-nearest-even
__device__ __forceinline__ unsigned short f2bf(float f) {
    unsigned int u = __float_as_uint(f);
    u = (u + 0x7FFFu + ((u >> 16) & 1u));
    return (unsigned short)(u >> 16);
}

__device__ __forceinline__ float silu(float v) {
    return v / (1.0f + __expf(-v));
}

// Gather a 16x32 bf16 A-fragment from LDS (row-major ushort, given row stride).
// ISA layout (16-bit A 16x32): lanes 0-15 / 16-31 both hold M=lane&15;
//   VGPR pair p in 0..3: K = khalf + 2p + o ; p in 4..7: K = 16 + khalf + 2(p-4) + o
//   khalf = (lane>=16) ? 8 : 0
__device__ __forceinline__ v16bf gather_a(const unsigned short* base, int stride, int kbase) {
    const int L = threadIdx.x & 31;
    const int m = L & 15;
    const int khalf = (L >> 4) * 8;
    v16us t;
#pragma unroll
    for (int e = 0; e < 16; ++e) {
        const int p = e >> 1, o = e & 1;
        const int k = (p < 4) ? (khalf + 2 * p + o) : (16 + khalf + 2 * (p - 4) + o);
        t[e] = base[m * stride + kbase + k];
    }
    return __builtin_bit_cast(v16bf, t);
}

// Load a pre-packed B-fragment (32 lanes x 16 bf16, lane-contiguous 32B).
__device__ __forceinline__ v16bf load_b(const unsigned short* p) {
    const int L = threadIdx.x & 31;
    v16us t = *(const v16us*)(p + (size_t)L * 16);
    return __builtin_bit_cast(v16bf, t);
}

// ---------------------------------------------------------------------------
// Weight packer: src[K x N] f32 row-major -> bf16 WMMA B-fragments.
// Tile (kt, nt): 512 ushorts, lane L holds K = (e<8) ? (L>>4)*8 + e : 16 + (L>>4)*8 + (e-8), N = nt*16 + (L&15).
// kreal = 32 (dense) or 16 (zero-pad K 16..31 for K=16 matrices).
// ---------------------------------------------------------------------------
__global__ void mamba_pack_b(const float* __restrict__ src, unsigned short* __restrict__ dst,
                             int N, int kreal, int total) {
    int idx = blockIdx.x * blockDim.x + threadIdx.x;
    if (idx >= total) return;
    const int tile   = idx >> 9;
    const int within = idx & 511;
    const int L = within >> 4;
    const int e = within & 15;
    const int NT = N >> 4;
    const int kt = tile / NT, nt = tile % NT;
    const int n = nt * 16 + (L & 15);
    const int klocal = (e < 8) ? ((L >> 4) * 8 + e) : (16 + (L >> 4) * 8 + (e - 8));
    unsigned short v = 0;
    if (klocal < kreal) {
        const int kg = kt * kreal + klocal;
        v = f2bf(src[(size_t)kg * N + n]);
    }
    dst[idx] = v;
}

// ---------------------------------------------------------------------------
// Scan prep: A^1..A^8 and r_k = ones @ A^k (k=0..7).  One block, 256 threads.
// ---------------------------------------------------------------------------
__global__ void mamba_scan_prep(const float* __restrict__ A,
                                float* __restrict__ Apow,   // [8][16][16], Apow[p-1] = A^p
                                float* __restrict__ rvec) { // [8][16]
    __shared__ float P[256], Q[256];
    const int i = threadIdx.x >> 4, j = threadIdx.x & 15;
    P[i * 16 + j] = A[i * 16 + j];
    __syncthreads();
    Apow[i * 16 + j] = P[i * 16 + j];          // A^1
    if (i == 0) rvec[j] = 1.0f;                // r_0 = ones
    for (int k = 2; k <= 8; ++k) {
        float s = 0.f;
#pragma unroll
        for (int m = 0; m < 16; ++m) s += P[i * 16 + m] * A[m * 16 + j];
        Q[i * 16 + j] = s;
        __syncthreads();
        P[i * 16 + j] = Q[i * 16 + j];
        __syncthreads();
        Apow[(k - 1) * 256 + i * 16 + j] = P[i * 16 + j];
    }
    if (i == 0) {
#pragma unroll
        for (int k = 1; k < 8; ++k) {
            float s = 0.f;
            for (int m = 0; m < 16; ++m) s += Apow[(k - 1) * 256 + m * 16 + j];
            rvec[k * 16 + j] = s;
        }
    }
}

// ---------------------------------------------------------------------------
// Kernel 1: fused LN -> GEMM1(512x1024,bf16 WMMA) -> SiLU -> GEMM2(->16) -> drive
// 32 tokens / block, 256 threads (8 waves). Writes drive[NTOK] only.
// ---------------------------------------------------------------------------
__global__ __launch_bounds__(256) void mamba_fused_front(
    const float* __restrict__ x, const float* __restrict__ ln_g, const float* __restrict__ ln_b,
    const unsigned short* __restrict__ pWin, const float* __restrict__ b_in,
    const unsigned short* __restrict__ pWssmIn, const float* __restrict__ b_ssm_in,
    const float* __restrict__ Bp, float* __restrict__ drive) {
    __shared__ unsigned short xln[32][512];       // LN'd tile, bf16 bits  (32 KB)
    __shared__ unsigned short wstage[8][16 * 32]; // per-wave silu staging (8 KB)
    __shared__ float uLds[32 * 16];               // u accumulator         (2 KB)

    const int tid = threadIdx.x;
    const int w = tid >> 5;
    const int L = tid & 31;
    const long tokBase = (long)blockIdx.x * 32;

    // init u with b_ssm_in
    uLds[tid] = b_ssm_in[tid & 15];
    uLds[tid + 256] = b_ssm_in[tid & 15];

    // ---- LayerNorm: wave w handles tokens 4w..4w+3, 16 elems per lane ----
#pragma unroll
    for (int tt = 0; tt < 4; ++tt) {
        const int tok = w * 4 + tt;
        const float* xr = x + (tokBase + tok) * HID;
        float s = 0.f, ss = 0.f, vals[16];
#pragma unroll
        for (int i = 0; i < 16; ++i) {
            float v = xr[L + 32 * i];
            vals[i] = v; s += v; ss += v * v;
        }
#pragma unroll
        for (int o = 16; o >= 1; o >>= 1) {
            s  += __shfl_xor(s,  o, 32);
            ss += __shfl_xor(ss, o, 32);
        }
        const float mu   = s * (1.f / 512.f);
        const float var  = ss * (1.f / 512.f) - mu * mu;
        const float rstd = rsqrtf(var + 1e-5f);
#pragma unroll
        for (int i = 0; i < 16; ++i) {
            const int col = L + 32 * i;
            xln[tok][col] = f2bf((vals[i] - mu) * rstd * ln_g[col] + ln_b[col]);
        }
    }
    __syncthreads();

    // ---- wave tiling: m_sub = token half, g = E-column quarter ----
    const int m_sub = w & 1;
    const int g = w >> 1;

    // Hoist all 16 A-fragments of this token sub-tile (K = 512 = 16 x 32)
    v16bf afrag[16];
#pragma unroll
    for (int kt = 0; kt < 16; ++kt)
        afrag[kt] = gather_a(&xln[m_sub * 16][0], 512, kt * 32);

    v8f ufrag = {};
    for (int ntl = 0; ntl < 16; ++ntl) {
        const int nt = g * 16 + ntl;
        v8f c = {};
#pragma unroll
        for (int kt = 0; kt < 16; ++kt) {
            v16bf b = load_b(pWin + (size_t)(kt * 64 + nt) * 512);
            c = __builtin_amdgcn_wmma_f32_16x16x32_bf16(false, afrag[kt], false, b,
                                                        (short)0, c, false, false);
        }
        if (ntl + 1 < 16)
            __builtin_prefetch(pWin + (size_t)(g * 16 + ntl + 1) * 512, 0, 0);

        const float bias = b_in[nt * 16 + (L & 15)];
        const int scol = (ntl & 1) * 16 + (L & 15);
#pragma unroll
        for (int r = 0; r < 8; ++r) {
            const int m = r + 8 * (L >> 4);
            wstage[w][m * 32 + scol] = f2bf(silu(c[r] + bias));
        }
        if (ntl & 1) { // a full K=32 chunk of E is staged -> accumulate u
            asm volatile("s_wait_dscnt 0" ::: "memory");
            v16bf a2 = gather_a(&wstage[w][0], 32, 0);
            v16bf b2 = load_b(pWssmIn + (size_t)(nt >> 1) * 512);
            ufrag = __builtin_amdgcn_wmma_f32_16x16x32_bf16(false, a2, false, b2,
                                                            (short)0, ufrag, false, false);
        }
    }

    // reduce partial u across the 4 waves of each m_sub
#pragma unroll
    for (int r = 0; r < 8; ++r) {
        const int m = r + 8 * (L >> 4);
        atomicAdd(&uLds[(m_sub * 16 + m) * 16 + (L & 15)], ufrag[r]);
    }
    __syncthreads();

    // drive = u @ Bp  (per-token scalar)
    if (tid < 32) {
        float acc = 0.f;
#pragma unroll
        for (int sct = 0; sct < 16; ++sct) acc += uLds[tid * 16 + sct] * Bp[sct];
        drive[tokBase + tid] = acc;
    }
}

// ---------------------------------------------------------------------------
// Kernel 2: 8x-unrolled sequential scan. 1 wave per batch.
// h_{t0+k} = H @ A^{k+1} + sum_{m<=k} d_{t0+m} * r_{k-m};  ys = h * C  (bf16)
// ---------------------------------------------------------------------------
__global__ __launch_bounds__(32) void mamba_scan(
    const float* __restrict__ drive, const float* __restrict__ Apow,
    const float* __restrict__ rvec, const float* __restrict__ C,
    unsigned short* __restrict__ ys) {
    const int b = blockIdx.x;
    const int L = threadIdx.x;
    const int j = L & 15;

    float Acol[8][16];   // columns j of A^1..A^8
#pragma unroll
    for (int k = 0; k < 8; ++k)
#pragma unroll
        for (int i = 0; i < 16; ++i)
            Acol[k][i] = Apow[k * 256 + i * 16 + j];

    float rv[8];
#pragma unroll
    for (int k = 0; k < 8; ++k) rv[k] = rvec[k * 16 + j];
    const float Cj = C[j];

    float h[16];
#pragma unroll
    for (int i = 0; i < 16; ++i) h[i] = 0.f;

    const float* dptr = drive + (size_t)b * SEQ;
    unsigned short* yb = ys + (size_t)b * SEQ * STATE;

    for (int t0 = 0; t0 < SEQ; t0 += 8) {
        float d[8];
#pragma unroll
        for (int m = 0; m < 8; ++m) d[m] = dptr[t0 + m];

        float out7 = 0.f;
#pragma unroll
        for (int k = 0; k < 8; ++k) {
            float s = 0.f;
#pragma unroll
            for (int i = 0; i < 16; ++i) s += h[i] * Acol[k][i];
#pragma unroll
            for (int m = 0; m <= k; ++m) s += d[m] * rv[k - m];
            if (L < 16) yb[(size_t)(t0 + k) * STATE + j] = f2bf(s * Cj);
            if (k == 7) out7 = s;
        }
#pragma unroll
        for (int i = 0; i < 16; ++i) h[i] = __shfl(out7, i, 32);
    }
}

// ---------------------------------------------------------------------------
// Kernel 3: fused GEMM3(ys@W_ssm_out, K=16 pad 32) -> SiLU -> GEMM4(@W_out)
//           -> + b_out + residual.  32 tokens / block, 256 threads (8 waves).
// ---------------------------------------------------------------------------
__global__ __launch_bounds__(256) void mamba_fused_back(
    const unsigned short* __restrict__ ys, const unsigned short* __restrict__ pWso,
    const float* __restrict__ b_ssm_out, const unsigned short* __restrict__ pWout,
    const float* __restrict__ b_out, const float* __restrict__ x,
    float* __restrict__ out) {
    __shared__ unsigned short ysA[32][32];        // K zero-padded to 32 (2 KB)
    __shared__ unsigned short wstage[8][16 * 32]; // per-wave silu staging (8 KB)

    const int tid = threadIdx.x;
    const int w = tid >> 5;
    const int L = tid & 31;
    const long tokBase = (long)blockIdx.x * 32;

    { // stage ys tile, zero-pad cols 16..31
        int r = tid >> 4, c0 = tid & 15;
        ysA[r][c0] = ys[(tokBase + r) * STATE + c0];
        ysA[r][16 + c0] = 0;
        r += 16;
        ysA[r][c0] = ys[(tokBase + r) * STATE + c0];
        ysA[r][16 + c0] = 0;
    }
    __syncthreads();

    const int m_sub = w & 1;  // token half
    const int q = w >> 1;     // output-column quarter (8 of 32 n-tiles)

    const v16bf aYs = gather_a(&ysA[m_sub * 16][0], 32, 0);

    v8f acc[8] = {};
    for (int c = 0; c < 32; ++c) { // K-chunks of 32 over E = 1024
#pragma unroll
        for (int half = 0; half < 2; ++half) {
            const int e = 2 * c + half;
            v16bf bso = load_b(pWso + (size_t)e * 512);
            v8f t = {};
            t = __builtin_amdgcn_wmma_f32_16x16x32_bf16(false, aYs, false, bso,
                                                        (short)0, t, false, false);
            const float bias = b_ssm_out[e * 16 + (L & 15)];
#pragma unroll
            for (int r = 0; r < 8; ++r) {
                const int m = r + 8 * (L >> 4);
                wstage[w][m * 32 + half * 16 + (L & 15)] = f2bf(silu(t[r] + bias));
            }
        }
        asm volatile("s_wait_dscnt 0" ::: "memory");
        v16bf a2 = gather_a(&wstage[w][0], 32, 0);
        if (c + 1 < 32)
            __builtin_prefetch(pWout + (size_t)((c + 1) * 32 + q * 8) * 512, 0, 0);
#pragma unroll
        for (int i = 0; i < 8; ++i) {
            const int nt = q * 8 + i;
            v16bf bw = load_b(pWout + (size_t)(c * 32 + nt) * 512);
            acc[i] = __builtin_amdgcn_wmma_f32_16x16x32_bf16(false, a2, false, bw,
                                                             (short)0, acc[i], false, false);
        }
    }

    // epilogue: + b_out + residual
#pragma unroll
    for (int i = 0; i < 8; ++i) {
        const int nt = q * 8 + i;
        const int col = nt * 16 + (L & 15);
        const float bo = b_out[col];
#pragma unroll
        for (int r = 0; r < 8; ++r) {
            const int m = r + 8 * (L >> 4);
            const long tok = tokBase + m_sub * 16 + m;
            out[tok * HID + col] = acc[i][r] + bo + x[tok * HID + col];
        }
    }
}

// ---------------------------------------------------------------------------
// Host launcher
// ---------------------------------------------------------------------------
extern "C" void kernel_launch(void* const* d_in, const int* in_sizes, int n_in,
                              void* d_out, int out_size, void* d_ws, size_t ws_size,
                              hipStream_t stream) {
    const float* x         = (const float*)d_in[0];
    const float* ln_g      = (const float*)d_in[1];
    const float* ln_b      = (const float*)d_in[2];
    const float* W_in      = (const float*)d_in[3];
    const float* b_in      = (const float*)d_in[4];
    const float* W_ssm_in  = (const float*)d_in[5];
    const float* b_ssm_in  = (const float*)d_in[6];
    const float* A         = (const float*)d_in[7];
    const float* Bp        = (const float*)d_in[8];
    const float* C         = (const float*)d_in[9];
    const float* W_ssm_out = (const float*)d_in[10];
    const float* b_ssm_out = (const float*)d_in[11];
    const float* W_out     = (const float*)d_in[12];
    const float* b_out     = (const float*)d_in[13];
    float* outp = (float*)d_out;

    // workspace carve-up (all chunks 256B aligned)
    char* ws = (char*)d_ws;
    float* drive = (float*)ws;              ws += (size_t)NTOK * 4;             // 256 KB
    unsigned short* ysb = (unsigned short*)ws; ws += (size_t)NTOK * STATE * 2;  // 2 MB
    float* Apow = (float*)ws;               ws += 8 * 256 * 4;                  // 8 KB
    float* rvec = (float*)ws;               ws += 1024;
    unsigned short* pWin    = (unsigned short*)ws; ws += (size_t)16 * 64 * 512 * 2; // 1 MB
    unsigned short* pWssmIn = (unsigned short*)ws; ws += (size_t)32 * 1  * 512 * 2;
    unsigned short* pWso    = (unsigned short*)ws; ws += (size_t)1  * 64 * 512 * 2;
    unsigned short* pWout   = (unsigned short*)ws; ws += (size_t)32 * 32 * 512 * 2; // 1 MB

    // pack weights into WMMA B-fragment layout (bf16)
    mamba_pack_b<<<2048, 256, 0, stream>>>(W_in,      pWin,    EXP,  32, 16 * 64 * 512);
    mamba_pack_b<<<64,   256, 0, stream>>>(W_ssm_in,  pWssmIn, STATE, 32, 32 * 1 * 512);
    mamba_pack_b<<<128,  256, 0, stream>>>(W_ssm_out, pWso,    EXP,  16, 1 * 64 * 512);
    mamba_pack_b<<<2048, 256, 0, stream>>>(W_out,     pWout,   HID,  32, 32 * 32 * 512);

    mamba_scan_prep<<<1, 256, 0, stream>>>(A, Apow, rvec);

    mamba_fused_front<<<NTOK / 32, 256, 0, stream>>>(x, ln_g, ln_b, pWin, b_in,
                                                     pWssmIn, b_ssm_in, Bp, drive);

    mamba_scan<<<BATCH, 32, 0, stream>>>(drive, Apow, rvec, C, ysb);

    mamba_fused_back<<<NTOK / 32, 256, 0, stream>>>(ysb, pWso, b_ssm_out, pWout,
                                                    b_out, x, outp);
}
